// NodeModel_32478542693150
// MI455X (gfx1250) — compile-verified, matrix operand order checked by reading, not run on hardware
//
#include <hip/hip_runtime.h>

typedef __attribute__((ext_vector_type(16))) __bf16 bf16x16;
typedef __attribute__((ext_vector_type(8)))  __bf16 bf16x8;
typedef __attribute__((ext_vector_type(8)))  float  f32x8;
typedef __attribute__((ext_vector_type(4)))  int    i32x4;

#if __has_builtin(__builtin_amdgcn_global_load_async_to_lds_b128) && \
    __has_builtin(__builtin_amdgcn_s_wait_asynccnt)
#define USE_ASYNC 1
#else
#define USE_ASYNC 0
#endif

#define WMMA_BF16(a, b, c) \
  __builtin_amdgcn_wmma_f32_16x16x32_bf16(false, (a), false, (b), (short)0, (c), false, false)

// ---------------------------------------------------------------------------
// Weight pre-pack: fp32 [K,N] row-major -> bf16 WMMA B-fragment order.
// Fragment layout: Wp[((c*ntiles + t)*32 + lane)*16 + v], where lane = 16h+n,
// K = 32c + ((v&8)<<1) + (v&7) + 8h, Ncol = 16t + n  (CDNA5 16-bit 16x32 layout)
// ---------------------------------------------------------------------------
__device__ __forceinline__ void pack_one(const float* __restrict__ W,
                                         __bf16* __restrict__ Wp,
                                         int id, int ntiles, int ncols) {
  const int v  = id & 15;
  const int L  = (id >> 4) & 31;
  const int tc = id >> 9;            // t + c*ntiles
  const int t  = tc % ntiles;
  const int c  = tc / ntiles;
  const int h  = L >> 4;
  const int n  = L & 15;
  const int K  = 32 * c + ((v & 8) << 1) + (v & 7) + 8 * h;
  Wp[id] = (__bf16)W[K * ncols + 16 * t + n];
}

__global__ void prep_weights(const float* __restrict__ W1, const float* __restrict__ W2,
                             const float* __restrict__ W3, const float* __restrict__ W4,
                             __bf16* __restrict__ W1p, __bf16* __restrict__ W2p,
                             __bf16* __restrict__ W3p, __bf16* __restrict__ W4p) {
  const int id = blockIdx.x * blockDim.x + threadIdx.x;
  if (id < 65536)                   pack_one(W1, W1p, id,           16, 256);
  else if (id < 131072)             pack_one(W2, W2p, id - 65536,   16, 256);
  else if (id < 245760)             pack_one(W3, W3p, id - 131072,  16, 256);
  else if (id < 278528)             pack_one(W4, W4p, id - 245760,   8, 128);
}

__global__ void zero_agg(float* __restrict__ p, long long count) {
  for (long long i = (long long)blockIdx.x * blockDim.x + threadIdx.x; i < count;
       i += (long long)gridDim.x * blockDim.x)
    p[i] = 0.0f;
}

// ---------------------------------------------------------------------------
// Stage one weight chunk (bytes multiple of 2048) into LDS.
// Async path: GLOBAL_LOAD_ASYNC_TO_LDS_B128 (ASYNCcnt), overlapped with WMMA.
// ---------------------------------------------------------------------------
__device__ __forceinline__ void stage_B(const __bf16* __restrict__ src,
                                        __bf16* dst, int bytes, int tid) {
#if USE_ASYNC
#pragma unroll
  for (int off = tid * 16; off < bytes; off += 128 * 16)
    __builtin_amdgcn_global_load_async_to_lds_b128(
        (i32x4*)((char*)(void*)(__bf16*)src + off),
        (i32x4*)((char*)dst + off), 0, 0);
#else
#pragma unroll
  for (int off = tid * 16; off < bytes; off += 128 * 16)
    *(uint4*)((char*)dst + off) = *(const uint4*)((const char*)src + off);
#endif
}

__device__ __forceinline__ void stage_wait() {
#if USE_ASYNC
  __builtin_amdgcn_s_wait_asynccnt(0);
#endif
}

// Build a 16x32 bf16 A fragment for this lane from two contiguous 8-float runs
__device__ __forceinline__ bf16x16 make_a(float4 f0, float4 f1, float4 f2, float4 f3, float s) {
  __bf16 t[16] __attribute__((aligned(32)));
  t[0]  = (__bf16)(f0.x * s); t[1]  = (__bf16)(f0.y * s);
  t[2]  = (__bf16)(f0.z * s); t[3]  = (__bf16)(f0.w * s);
  t[4]  = (__bf16)(f1.x * s); t[5]  = (__bf16)(f1.y * s);
  t[6]  = (__bf16)(f1.z * s); t[7]  = (__bf16)(f1.w * s);
  t[8]  = (__bf16)(f2.x * s); t[9]  = (__bf16)(f2.y * s);
  t[10] = (__bf16)(f2.z * s); t[11] = (__bf16)(f2.w * s);
  t[12] = (__bf16)(f3.x * s); t[13] = (__bf16)(f3.y * s);
  t[14] = (__bf16)(f3.z * s); t[15] = (__bf16)(f3.w * s);
  return *(const bf16x16*)t;
}

#define SHUF16(lo, hi) __builtin_shufflevector(lo, hi, 0, 1, 2, 3, 4, 5, 6, 7, \
                                               8, 9, 10, 11, 12, 13, 14, 15)

// ---------------------------------------------------------------------------
// Edge MLP + fused scatter-sum.  Each wave: M=32 edges x N=128 cols.
// Wave pair (nh=0/1) covers a 32x256 tile; block = 2 pairs = 64 edges.
// ---------------------------------------------------------------------------
__global__ __launch_bounds__(128) void edge_kernel(
    const float* __restrict__ x, const long long* __restrict__ ei,
    const float* __restrict__ ea,
    const __bf16* __restrict__ W1p, const float* __restrict__ b1,
    const __bf16* __restrict__ W2p, const float* __restrict__ b2,
    float* __restrict__ aggsum, float* __restrict__ cnt, long long E) {
  __shared__ __align__(16) __bf16 sB[2][8192];   // 2 x 16KB weight double buffer
  __shared__ __align__(16) __bf16 sT[2 * 32 * 256]; // 2 groups x 32 edges x 256
  const int tid  = threadIdx.x;
  const int lane = tid & 31;
  const int wave = tid >> 5;
  const int pair = wave >> 1;     // edge group within block
  const int nh   = wave & 1;      // N-half handled by this wave
  const int h    = lane >> 4;
  const int n    = lane & 15;
  const long long gbase = (long long)blockIdx.x * 64 + pair * 32;

  long long e0 = gbase + n;       if (e0 >= E) e0 = E - 1;
  long long e1 = gbase + 16 + n;  if (e1 >= E) e1 = E - 1;
  const float* x0  = x + ei[e0] * 128;
  const float* x1  = x + ei[e1] * 128;
  const float* ea0 = ea + e0 * 128;
  const float* ea1 = ea + e1 * 128;

  // ---- Layer 1 ----
  stage_B(W1p, sB[0], 16384, tid);
  stage_wait();
  __syncthreads();

  float4 p00 = *(const float4*)(x0 + 8 * h),      p01 = *(const float4*)(x0 + 8 * h + 4);
  float4 p02 = *(const float4*)(x0 + 16 + 8 * h), p03 = *(const float4*)(x0 + 20 + 8 * h);
  float4 p10 = *(const float4*)(x1 + 8 * h),      p11 = *(const float4*)(x1 + 8 * h + 4);
  float4 p12 = *(const float4*)(x1 + 16 + 8 * h), p13 = *(const float4*)(x1 + 20 + 8 * h);

  f32x8 acc0[8] = {}, acc1[8] = {};
  for (int c = 0; c < 8; ++c) {
    const bf16x16 a0 = make_a(p00, p01, p02, p03, 1.0f);
    const bf16x16 a1 = make_a(p10, p11, p12, p13, 1.0f);
    if (c < 7) {
      const int cc = c + 1;
      const float* s0 = (cc < 4) ? (x0 + 32 * cc) : (ea0 + (32 * cc - 128));
      const float* s1 = (cc < 4) ? (x1 + 32 * cc) : (ea1 + (32 * cc - 128));
      p00 = *(const float4*)(s0 + 8 * h);      p01 = *(const float4*)(s0 + 8 * h + 4);
      p02 = *(const float4*)(s0 + 16 + 8 * h); p03 = *(const float4*)(s0 + 20 + 8 * h);
      p10 = *(const float4*)(s1 + 8 * h);      p11 = *(const float4*)(s1 + 8 * h + 4);
      p12 = *(const float4*)(s1 + 16 + 8 * h); p13 = *(const float4*)(s1 + 20 + 8 * h);
      stage_B(W1p + cc * 8192, sB[cc & 1], 16384, tid);   // overlap with WMMAs
    }
    const __bf16* bb = sB[c & 1] + nh * 4096;   // this wave's 8 N-tiles
#pragma unroll
    for (int t = 0; t < 8; ++t) {
      const bf16x16 bm = *(const bf16x16*)(bb + (t * 32 + lane) * 16);
      acc0[t] = WMMA_BF16(a0, bm, acc0[t]);     // B fragment reused twice
      acc1[t] = WMMA_BF16(a1, bm, acc1[t]);
    }
    stage_wait();
    __syncthreads();
  }

  // bias + ReLU -> transpose buffer (group-local 32x256 tile)
  const int gb = pair * 8192;
#pragma unroll
  for (int t = 0; t < 8; ++t) {
    const int col = (nh * 8 + t) * 16 + n;
    const float bias = b1[col];
#pragma unroll
    for (int r = 0; r < 8; ++r) {
      sT[gb + (r + 8 * h) * 256 + col]      = (__bf16)fmaxf(acc0[t][r] + bias, 0.0f);
      sT[gb + (16 + r + 8 * h) * 256 + col] = (__bf16)fmaxf(acc1[t][r] + bias, 0.0f);
    }
  }

  // ---- Layer 2 ----
  stage_B(W2p, sB[0], 16384, tid);
  stage_wait();
  __syncthreads();   // covers sT writes (cross-wave) and sB[0] staging

  f32x8 acc20[8] = {}, acc21[8] = {};
  for (int c = 0; c < 8; ++c) {
    const bf16x8 lo0 = *(const bf16x8*)&sT[gb + n * 256 + 32 * c + 8 * h];
    const bf16x8 hi0 = *(const bf16x8*)&sT[gb + n * 256 + 32 * c + 16 + 8 * h];
    const bf16x8 lo1 = *(const bf16x8*)&sT[gb + (16 + n) * 256 + 32 * c + 8 * h];
    const bf16x8 hi1 = *(const bf16x8*)&sT[gb + (16 + n) * 256 + 32 * c + 16 + 8 * h];
    const bf16x16 a0 = SHUF16(lo0, hi0);
    const bf16x16 a1 = SHUF16(lo1, hi1);
    if (c < 7) stage_B(W2p + (c + 1) * 8192, sB[(c + 1) & 1], 16384, tid);
    const __bf16* bb = sB[c & 1] + nh * 4096;
#pragma unroll
    for (int t = 0; t < 8; ++t) {
      const bf16x16 bm = *(const bf16x16*)(bb + (t * 32 + lane) * 16);
      acc20[t] = WMMA_BF16(a0, bm, acc20[t]);
      acc21[t] = WMMA_BF16(a1, bm, acc21[t]);
    }
    stage_wait();
    __syncthreads();
  }

  // ---- fused scatter: agg_sum[col[e]] += h_e, cnt[col[e]] += 1 ----
#pragma unroll
  for (int i = 0; i < 2; ++i) {
#pragma unroll
    for (int r = 0; r < 8; ++r) {
      const long long e = gbase + i * 16 + r + 8 * h;
      if (e < E) {
        float* aggrow = aggsum + ei[E + e] * 256;
#pragma unroll
        for (int t = 0; t < 8; ++t) {
          const int col = (nh * 8 + t) * 16 + n;
          const float v = (i == 0 ? acc20[t][r] : acc21[t][r]) + b2[col];
          atomicAdd(aggrow + col, v);
        }
      }
    }
  }
  if (nh == 0) {   // one wave per pair counts its 32 edges
    const long long e = gbase + lane;
    if (e < E) atomicAdd(cnt + ei[E + e], 1.0f);
  }
}

// ---------------------------------------------------------------------------
// Node MLP: out = ReLU([x || agg_mean || u[batch]] @ W3 + b3) @ W4 + b4
// Same M=32 x N-half tiling.
// ---------------------------------------------------------------------------
__global__ __launch_bounds__(128) void node_kernel(
    const float* __restrict__ x, const float* __restrict__ aggsum,
    const float* __restrict__ cnt, const float* __restrict__ u,
    const long long* __restrict__ batch,
    const __bf16* __restrict__ W3p, const float* __restrict__ b3,
    const __bf16* __restrict__ W4p, const float* __restrict__ b4,
    float* __restrict__ out, long long Nn) {
  __shared__ __align__(16) __bf16 sB[2][8192];
  __shared__ __align__(16) __bf16 sT[2 * 32 * 256];
  const int tid  = threadIdx.x;
  const int lane = tid & 31;
  const int wave = tid >> 5;
  const int pair = wave >> 1;
  const int nh   = wave & 1;
  const int h    = lane >> 4;
  const int n    = lane & 15;
  const long long gbase = (long long)blockIdx.x * 64 + pair * 32;

  long long n0 = gbase + n;       if (n0 >= Nn) n0 = Nn - 1;
  long long n1 = gbase + 16 + n;  if (n1 >= Nn) n1 = Nn - 1;
  const float* x0 = x + n0 * 128;      const float* x1 = x + n1 * 128;
  const float* a0r = aggsum + n0 * 256; const float* a1r = aggsum + n1 * 256;
  const float* u0 = u + batch[n0] * 64; const float* u1 = u + batch[n1] * 64;
  const float rinv0 = 1.0f / fmaxf(cnt[n0], 1.0f);
  const float rinv1 = 1.0f / fmaxf(cnt[n1], 1.0f);

  // ---- Layer 1: A[32 x 448] @ W3[448 x 256] ----
  stage_B(W3p, sB[0], 16384, tid);
  stage_wait();
  __syncthreads();

  float4 p00 = *(const float4*)(x0 + 8 * h),      p01 = *(const float4*)(x0 + 8 * h + 4);
  float4 p02 = *(const float4*)(x0 + 16 + 8 * h), p03 = *(const float4*)(x0 + 20 + 8 * h);
  float4 p10 = *(const float4*)(x1 + 8 * h),      p11 = *(const float4*)(x1 + 8 * h + 4);
  float4 p12 = *(const float4*)(x1 + 16 + 8 * h), p13 = *(const float4*)(x1 + 20 + 8 * h);

  f32x8 acc0[8] = {}, acc1[8] = {};
  for (int c = 0; c < 14; ++c) {
    const bool isAgg = (c >= 4 && c < 12);
    const bf16x16 a0 = make_a(p00, p01, p02, p03, isAgg ? rinv0 : 1.0f);
    const bf16x16 a1 = make_a(p10, p11, p12, p13, isAgg ? rinv1 : 1.0f);
    if (c < 13) {
      const int cc = c + 1;
      const float *s0, *s1;
      if (cc < 4)       { s0 = x0 + 32 * cc;          s1 = x1 + 32 * cc; }
      else if (cc < 12) { s0 = a0r + (32 * cc - 128); s1 = a1r + (32 * cc - 128); }
      else              { s0 = u0 + (32 * cc - 384);  s1 = u1 + (32 * cc - 384); }
      p00 = *(const float4*)(s0 + 8 * h);      p01 = *(const float4*)(s0 + 8 * h + 4);
      p02 = *(const float4*)(s0 + 16 + 8 * h); p03 = *(const float4*)(s0 + 20 + 8 * h);
      p10 = *(const float4*)(s1 + 8 * h);      p11 = *(const float4*)(s1 + 8 * h + 4);
      p12 = *(const float4*)(s1 + 16 + 8 * h); p13 = *(const float4*)(s1 + 20 + 8 * h);
      stage_B(W3p + cc * 8192, sB[cc & 1], 16384, tid);
    }
    const __bf16* bb = sB[c & 1] + nh * 4096;
#pragma unroll
    for (int t = 0; t < 8; ++t) {
      const bf16x16 bm = *(const bf16x16*)(bb + (t * 32 + lane) * 16);
      acc0[t] = WMMA_BF16(a0, bm, acc0[t]);
      acc1[t] = WMMA_BF16(a1, bm, acc1[t]);
    }
    stage_wait();
    __syncthreads();
  }

  const int gb = pair * 8192;
#pragma unroll
  for (int t = 0; t < 8; ++t) {
    const int col = (nh * 8 + t) * 16 + n;
    const float bias = b3[col];
#pragma unroll
    for (int r = 0; r < 8; ++r) {
      sT[gb + (r + 8 * h) * 256 + col]      = (__bf16)fmaxf(acc0[t][r] + bias, 0.0f);
      sT[gb + (16 + r + 8 * h) * 256 + col] = (__bf16)fmaxf(acc1[t][r] + bias, 0.0f);
    }
  }

  // ---- Layer 2: [32 x 128] = relu1[32 x 256] @ W4[256 x 128], 4 tiles/wave ----
  stage_B(W4p, sB[0], 8192, tid);
  stage_wait();
  __syncthreads();

  f32x8 acc20[4] = {}, acc21[4] = {};
  for (int c = 0; c < 8; ++c) {
    const bf16x8 lo0 = *(const bf16x8*)&sT[gb + n * 256 + 32 * c + 8 * h];
    const bf16x8 hi0 = *(const bf16x8*)&sT[gb + n * 256 + 32 * c + 16 + 8 * h];
    const bf16x8 lo1 = *(const bf16x8*)&sT[gb + (16 + n) * 256 + 32 * c + 8 * h];
    const bf16x8 hi1 = *(const bf16x8*)&sT[gb + (16 + n) * 256 + 32 * c + 16 + 8 * h];
    const bf16x16 a0 = SHUF16(lo0, hi0);
    const bf16x16 a1 = SHUF16(lo1, hi1);
    if (c < 7) stage_B(W4p + (c + 1) * 4096, sB[(c + 1) & 1], 8192, tid);
    const __bf16* bb = sB[c & 1] + nh * 2048;   // this wave's 4 N-tiles
#pragma unroll
    for (int t = 0; t < 4; ++t) {
      const bf16x16 bm = *(const bf16x16*)(bb + (t * 32 + lane) * 16);
      acc20[t] = WMMA_BF16(a0, bm, acc20[t]);
      acc21[t] = WMMA_BF16(a1, bm, acc21[t]);
    }
    stage_wait();
    __syncthreads();
  }

#pragma unroll
  for (int i = 0; i < 2; ++i) {
#pragma unroll
    for (int r = 0; r < 8; ++r) {
      const long long nr = gbase + i * 16 + r + 8 * h;
      if (nr < Nn) {
#pragma unroll
        for (int t = 0; t < 4; ++t) {
          const int col = (nh * 4 + t) * 16 + n;
          out[nr * 128 + col] = (i == 0 ? acc20[t][r] : acc21[t][r]) + b4[col];
        }
      }
    }
  }
}

// ---------------------------------------------------------------------------
extern "C" void kernel_launch(void* const* d_in, const int* in_sizes, int n_in,
                              void* d_out, int out_size, void* d_ws, size_t ws_size,
                              hipStream_t stream) {
  const float*     x     = (const float*)d_in[0];
  const long long* ei    = (const long long*)d_in[1];   // (2, E) int64
  const float*     ea    = (const float*)d_in[2];
  const float*     u     = (const float*)d_in[3];
  const long long* batch = (const long long*)d_in[4];
  const float* W1 = (const float*)d_in[5];  const float* b1 = (const float*)d_in[6];
  const float* W2 = (const float*)d_in[7];  const float* b2 = (const float*)d_in[8];
  const float* W3 = (const float*)d_in[9];  const float* b3 = (const float*)d_in[10];
  const float* W4 = (const float*)d_in[11]; const float* b4 = (const float*)d_in[12];

  const long long Nn = in_sizes[0] / 128;   // nodes
  const long long E  = in_sizes[2] / 128;   // edges
  float* out = (float*)d_out;

  // Workspace layout
  char* ws = (char*)d_ws;
  __bf16* W1p = (__bf16*)ws;                 // 65536 elems
  __bf16* W2p = W1p + 65536;                 // 65536
  __bf16* W3p = W2p + 65536;                 // 114688
  __bf16* W4p = W3p + 114688;                // 32768
  float* aggsum = (float*)(ws + 278528 * 2); // [Nn, 256] f32
  float* cnt    = aggsum + Nn * 256;         // [Nn] f32 (contiguous after aggsum)

  prep_weights<<<(278528 + 255) / 256, 256, 0, stream>>>(W1, W2, W3, W4,
                                                         W1p, W2p, W3p, W4p);
  zero_agg<<<2048, 256, 0, stream>>>(aggsum, Nn * 256 + Nn);

  const int eblocks = (int)((E + 63) / 64);
  edge_kernel<<<eblocks, 128, 0, stream>>>(x, ei, ea, W1p, b1, W2p, b2,
                                           aggsum, cnt, E);

  const int nblocks = (int)((Nn + 63) / 64);
  node_kernel<<<nblocks, 128, 0, stream>>>(x, aggsum, cnt, u, batch,
                                           W3p, b3, W4p, b4, out, Nn);
}
